// Flow1_7464653160938
// MI455X (gfx1250) — matrix operand order repaired and checked by reading, not compiled, not run on hardware
//
#include <hip/hip_runtime.h>

// ---------------------------------------------------------------------------
// Flow kernel for MI455X (gfx1250, wave32, WMMA).
//   B=262144, Z=128, ZH=64, HS=50 (padded to 64), NF=2 -> 4 couplings.
// Each wave handles 16 batch rows; matmuls on v_wmma_f32_16x16x32_bf16.
// All exp/log run as native exp2/log2 with constants folded into prepped
// weights: wsB = 2*log2e*b_in (+1e9 at col 50 => h[50]=1 carries mu/sig
// biases in weight row k=50); W_sig is prescaled by -log2e.
// ---------------------------------------------------------------------------

typedef __attribute__((ext_vector_type(16))) __bf16 v16bf;
typedef __attribute__((ext_vector_type(8)))  __bf16 v8bf;
typedef __attribute__((ext_vector_type(8)))  float  v8f;

#define FLOW_B   262144
#define WSW_ELEMS (4*3*8*32*16)        // 49152 bf16 fragment elements
#define WSW_BYTES (WSW_ELEMS*2)        // 98304 bytes
#define LOG2E     1.44269504088896f
#define LN2       0.69314718055995f
#define WMMA_BF16(A,Bm,C) \
  __builtin_amdgcn_wmma_f32_16x16x32_bf16(false,(A),false,(Bm),(short)0,(C),false,false)

__device__ __forceinline__ v16bf cat16(v8bf lo, v8bf hi) {
  return __builtin_shufflevector(lo, hi, 0,1,2,3,4,5,6,7,8,9,10,11,12,13,14,15);
}

// B-fragment: prepacked so each lane's 16 bf16 are contiguous (2x b128 loads)
__device__ __forceinline__ v16bf load_bfrag(const __bf16* p) {
  const v8bf* q = (const v8bf*)p;
  return cat16(q[0], q[1]);
}

// A-fragment from row-major LDS tile [16][72] bf16.
// Lane (half,m): elems 0..7 -> K = kt*32 + half*8 + e, elems 8..15 -> +16.
__device__ __forceinline__ v16bf load_afrag(const __bf16* row, int kt, int half) {
  const int k0 = kt*32 + half*8;
  v8bf lo = *(const v8bf*)(row + k0);
  v8bf hi = *(const v8bf*)(row + k0 + 16);
  return cat16(lo, hi);
}

// One affine coupling. zin is already staged (bf16) in this wave's LDS tile.
//   h = tanh(W0 zin + b0); h[50] saturates to 1.0 (bias 1e9) so weight row
//   k=50 of W1/W2 carries b_mu/b_sig.
//   zout = zout*sigmoid + mu ; lu2 += log2(1/sigmoid)  (W_sig pre * -log2e)
// If STAGE_OUT, zout is restaged into LDS as the next coupling's zin.
template <bool STAGE_OUT>
__device__ __forceinline__ void coupling(
    int c, const __bf16* __restrict__ wsW, const float* __restrict__ wsB,
    __bf16 (*ldsT)[72], float zout[4][8], float* lu2, int lane, int nl, int half)
{
  asm volatile("s_wait_dscnt 0" ::: "memory");
  const __bf16* rowp = &ldsT[nl][0];              // A-frag: M = lane&15
  v16bf a0 = load_afrag(rowp, 0, half);
  v16bf a1 = load_afrag(rowp, 1, half);
  const __bf16* wc = wsW + (size_t)(c*3) * (8*32*16);

  // ---- matmul 1: independent accumulator chains (inline C=0) -------------
  v8f acc[4];
#pragma unroll
  for (int nt = 0; nt < 4; ++nt) {
    v16bf b0 = load_bfrag(wc + ((size_t)((nt*2 + 0)*32 + lane))*16);
    v16bf b1 = load_bfrag(wc + ((size_t)((nt*2 + 1)*32 + lane))*16);
    v8f z = {};
    z = WMMA_BF16(a0, b0, z);
    acc[nt] = WMMA_BF16(a1, b1, z);
  }
  // ---- tanh(x) = 1 - 2/(exp2(2*log2e*x + bv2)+1); compute then store ------
  float th[4][8];
#pragma unroll
  for (int nt = 0; nt < 4; ++nt) {
    float bv2 = wsB[c*64 + nt*16 + nl];           // 2*log2e*b_in (1e9 @ col50)
#pragma unroll
    for (int j = 0; j < 8; ++j) {
      float e2 = __builtin_amdgcn_exp2f(__builtin_fmaf(acc[nt][j], 2.0f*LOG2E, bv2));
      th[nt][j] = 1.0f - 2.0f * __builtin_amdgcn_rcpf(e2 + 1.0f);
    }
  }
#pragma unroll
  for (int nt = 0; nt < 4; ++nt)
#pragma unroll
    for (int j = 0; j < 8; ++j)
      ldsT[half*8 + j][nl + 16*nt] = (__bf16)th[nt][j];

  asm volatile("s_wait_dscnt 0" ::: "memory");
  v16bf h0 = load_afrag(rowp, 0, half);
  v16bf h1 = load_afrag(rowp, 1, half);

  // ---- matmuls 2/3: 16 independent WMMAs, then one elementwise pass -------
  v8f am[4], as[4];
#pragma unroll
  for (int nt = 0; nt < 4; ++nt) {
    v16bf bm0 = load_bfrag(wc + ((size_t)((8  + nt*2 + 0)*32 + lane))*16);
    v16bf bm1 = load_bfrag(wc + ((size_t)((8  + nt*2 + 1)*32 + lane))*16);
    v16bf bs0 = load_bfrag(wc + ((size_t)((16 + nt*2 + 0)*32 + lane))*16);
    v16bf bs1 = load_bfrag(wc + ((size_t)((16 + nt*2 + 1)*32 + lane))*16);
    v8f m = {}; m = WMMA_BF16(h0, bm0, m); am[nt] = WMMA_BF16(h1, bm1, m);
    v8f s = {}; s = WMMA_BF16(h0, bs0, s); as[nt] = WMMA_BF16(h1, bs1, s);
  }
  // as = -log2e * (W_sig h + b_sig)  =>  u = 1/sigmoid = 1 + exp2(as)
#pragma unroll
  for (int nt = 0; nt < 4; ++nt) {
#pragma unroll
    for (int j = 0; j < 8; ++j) {
      float u  = 1.0f + __builtin_amdgcn_exp2f(as[nt][j]);
      float sg = __builtin_amdgcn_rcpf(u);
      zout[nt][j] = __builtin_fmaf(zout[nt][j], sg, am[nt][j]);
      lu2[j] += __builtin_amdgcn_logf(u);         // log2(u) = -log2(sigmoid)
    }
  }
  if (STAGE_OUT) {
#pragma unroll
    for (int nt = 0; nt < 4; ++nt)
#pragma unroll
      for (int j = 0; j < 8; ++j)
        ldsT[half*8 + j][nl + 16*nt] = (__bf16)zout[nt][j];
  }
}

// ---------------------------------------------------------------------------
// Weight prep: padded bf16 B-fragments; bias row k=50 for mu/sig matmuls;
// W_sig (and its bias row) prescaled by -log2e; wsB = 2*log2e*b_in with
// +1e9 at n=50 (tanh saturates to 1.0 -> bias carrier).
//   wsW[c][mat][frag(nt*2+kt)][lane][e]; n = nt*16+(lane&15),
//   k = kt*32 + (lane>>4)*8 + (e&7) + (e>>3)*16.
// ---------------------------------------------------------------------------
__global__ __launch_bounds__(256) void flow_prep(
    const float* __restrict__ W_in,  const float* __restrict__ b_in,
    const float* __restrict__ W_mu,  const float* __restrict__ b_mu,
    const float* __restrict__ W_sig, const float* __restrict__ b_sig,
    __bf16* __restrict__ wsW, float* __restrict__ wsB)
{
  int gid = blockIdx.x * 256 + threadIdx.x;
  if (gid < WSW_ELEMS) {
    int c   = gid / 12288;  int r  = gid % 12288;
    int mat = r / 4096;     int r2 = r % 4096;
    int frag = r2 / 512;    int le = r2 % 512;
    int L = le / 16, e = le % 16;
    int nt = frag >> 1, kt = frag & 1;
    int n = nt*16 + (L & 15);
    int k = kt*32 + (L >> 4)*8 + (e & 7) + (e >> 3)*16;
    float v = 0.0f;
    if (mat == 0) {                         // W_in: (c,HS=50,ZH=64), B[k=zh][n=hs]
      if (n < 50) v = W_in[(c*50 + n)*64 + k];
    } else {                                // W_mu/W_sig: (c,ZH=64,HS=50), B[k=hs][n=zh]
      const float* W  = (mat == 1) ? W_mu : W_sig;
      const float* bb = (mat == 1) ? b_mu : b_sig;
      if      (k < 50)  v = W[(c*64 + n)*50 + k];
      else if (k == 50) v = bb[c*64 + n];   // bias rides K row 50 (h[50]=1)
      if (mat == 2) v *= -LOG2E;            // fold exp2 conversion into W_sig
    }
    wsW[gid] = (__bf16)v;
  }
  if (gid < 4*64) {                         // wsB = 2*log2e*b_in, pad; 1e9@50
    int c = gid / 64, n = gid % 64;
    float v = 0.0f;
    if      (n < 50)  v = 2.0f*LOG2E * b_in[c*50 + n];
    else if (n == 50) v = 1.0e9f;           // tanh -> exactly 1.0
    wsB[gid] = v;
  }
}

// ---------------------------------------------------------------------------
// Main kernel: 256 threads = 8 waves, 16 rows/wave -> 128 rows/block.
// ---------------------------------------------------------------------------
__global__ __launch_bounds__(256) void flow_main(
    const float* __restrict__ mean, const float* __restrict__ logvar,
    const float* __restrict__ eps,
    const __bf16* __restrict__ wsW, const float* __restrict__ wsB,
    float* __restrict__ outz, float* __restrict__ outlpz,
    float* __restrict__ outlqz)
{
  __shared__ __bf16 lds[8][16][72];              // per-wave padded bf16 tiles
  const int tid  = threadIdx.x;
  const int lane = tid & 31;
  const int wv   = tid >> 5;
  const int nl   = lane & 15;
  const int half = lane >> 4;
  const long rowBase = ((long)blockIdx.x * 8 + wv) * 16;

  __bf16 (*ldsT)[72] = lds[wv];

  float z1t[4][8], z2t[4][8];
  float lq0[8], lu2[8];
#pragma unroll
  for (int j = 0; j < 8; ++j) { lq0[j] = 0.0f; lu2[j] = 0.0f; }

  // ---- load + reparameterize; stage z1 (bf16) into LDS as it is made ------
#pragma unroll
  for (int j = 0; j < 8; ++j) {
    const long r = rowBase + half*8 + j;
    const float* pm = mean   + r*128;
    const float* pl = logvar + r*128;
    const float* pe = eps    + r*128;
#pragma unroll
    for (int t = 0; t < 8; ++t) {
      const int col = nl + 16*t;
      float mn = __builtin_nontemporal_load(pm + col);
      float lv = __builtin_nontemporal_load(pl + col);
      float ep = __builtin_nontemporal_load(pe + col);
      float sd = __builtin_amdgcn_exp2f(0.5f*LOG2E * lv);  // exp(0.5*lv)
      float zv = __builtin_fmaf(ep, sd, mn);
      lq0[j] += __builtin_fmaf(ep, ep, lv);
      if (t < 4) {
        z1t[t][j] = zv;
        ldsT[half*8 + j][nl + 16*t] = (__bf16)zv;
      } else {
        z2t[t-4][j] = zv;
      }
    }
  }

  // ---- 4 couplings (zout of each is restaged as next zin) -----------------
  coupling<true >(0, wsW, wsB, ldsT, z2t, lu2, lane, nl, half);
  coupling<true >(1, wsW, wsB, ldsT, z1t, lu2, lane, nl, half);
  coupling<true >(2, wsW, wsB, ldsT, z2t, lu2, lane, nl, half);
  coupling<false>(3, wsW, wsB, ldsT, z1t, lu2, lane, nl, half);

  // ---- per-lane partials: q = ln2*lu2 - 0.5*lq0 ; lpz = sum z^2 -----------
  float q[8], lpz[8];
#pragma unroll
  for (int j = 0; j < 8; ++j) {
    q[j] = __builtin_fmaf(lq0[j], -0.5f, lu2[j]*LN2);
    float s = 0.0f;
#pragma unroll
    for (int t = 0; t < 4; ++t)
      s += z1t[t][j]*z1t[t][j] + z2t[t][j]*z2t[t][j];
    lpz[j] = s;
  }

  // ---- 16-lane butterfly reductions (masks 1,2,4,8 keep halves apart) -----
#pragma unroll
  for (int m = 1; m <= 8; m <<= 1) {
#pragma unroll
    for (int j = 0; j < 8; ++j) {
      q[j]   += __shfl_xor(q[j],   m, 32);
      lpz[j] += __shfl_xor(lpz[j], m, 32);
    }
  }

  // ---- store z (non-temporal; never re-read) ------------------------------
#pragma unroll
  for (int j = 0; j < 8; ++j) {
    const long r = rowBase + half*8 + j;
    float* po = outz + r*128;
#pragma unroll
    for (int t = 0; t < 4; ++t) {
      __builtin_nontemporal_store(z1t[t][j], po + nl + 16*t);
      __builtin_nontemporal_store(z2t[t][j], po + 64 + nl + 16*t);
    }
  }
  // ---- store logpz / logqz (one lane per row in each half) ----------------
#pragma unroll
  for (int j = 0; j < 8; ++j) {
    if (nl == j) {
      const long r = rowBase + half*8 + j;
      __builtin_nontemporal_store(-0.5f * lpz[j], outlpz + r);
      __builtin_nontemporal_store(q[j],           outlqz + r);
    }
  }
}

// ---------------------------------------------------------------------------
extern "C" void kernel_launch(void* const* d_in, const int* in_sizes, int n_in,
                              void* d_out, int out_size, void* d_ws, size_t ws_size,
                              hipStream_t stream) {
  const float* mean   = (const float*)d_in[0];
  const float* logvar = (const float*)d_in[1];
  const float* eps    = (const float*)d_in[2];
  const float* W_in   = (const float*)d_in[3];
  const float* b_in   = (const float*)d_in[4];
  const float* W_mu   = (const float*)d_in[5];
  const float* b_mu   = (const float*)d_in[6];
  const float* W_sig  = (const float*)d_in[7];
  const float* b_sig  = (const float*)d_in[8];

  __bf16* wsW = (__bf16*)d_ws;
  float*  wsB = (float*)((char*)d_ws + WSW_BYTES);

  float* outz   = (float*)d_out;
  float* outlpz = outz + (size_t)FLOW_B * 128;
  float* outlqz = outlpz + FLOW_B;

  flow_prep<<<WSW_ELEMS/256, 256, 0, stream>>>(W_in, b_in, W_mu, b_mu, W_sig, b_sig,
                                               wsW, wsB);
  flow_main<<<FLOW_B/128, 256, 0, stream>>>(mean, logvar, eps, wsW, wsB,
                                            outz, outlpz, outlqz);
}
